// BatchAllTripletLoss_86414741996347
// MI455X (gfx1250) — matrix operand order repaired and checked by reading, not compiled, not run on hardware
//
#include <hip/hip_runtime.h>
#include <hip/hip_bf16.h>
#include <math.h>

#define N 384
#define DIM 256
#define NT 24                 // 16-wide tiles per matrix dim (384/16)
#define MARGIN 0.3f
#define POS_TH 25.0f
#define NEG_TH 100.0f
#define R_EARTH 6371000.0f
#define DEG2RAD 0.017453292519943295f

typedef __attribute__((ext_vector_type(16))) _Float16 v16h;
typedef __attribute__((ext_vector_type(8)))  _Float16 v8h;
typedef __attribute__((ext_vector_type(8)))  float    v8f;

// ---------------------------------------------------------------------------
// Kernel 1: convert embeddings f32 -> f16 (row major) and compute row norms^2
// ---------------------------------------------------------------------------
__global__ void convert_norm_kernel(const float* __restrict__ E,
                                    _Float16* __restrict__ Eh,
                                    float* __restrict__ norms) {
  const int row = blockIdx.x;
  const int t   = threadIdx.x;           // 256 threads = one element each
  float v = E[row * DIM + t];
  Eh[row * DIM + t] = (_Float16)v;

  __shared__ float r[DIM];
  r[t] = v * v;
  __syncthreads();
  for (int s = DIM / 2; s > 0; s >>= 1) {
    if (t < s) r[t] += r[t + s];
    __syncthreads();
  }
  if (t == 0) norms[row] = r[0];
}

// ---------------------------------------------------------------------------
// Kernel 2: haversine GPS masks (pos: < 25m & i!=j ; neg: > 100m)
// ---------------------------------------------------------------------------
__global__ void mask_kernel(const float* __restrict__ gps,
                            unsigned char* __restrict__ pos,
                            unsigned char* __restrict__ neg) {
  const int i = blockIdx.x;
  const int j = threadIdx.x;             // 384 threads
  float a0 = gps[2 * i + 0], a1 = gps[2 * i + 1];
  float b0 = gps[2 * j + 0], b1 = gps[2 * j + 1];
  float lat1 = a0 * DEG2RAD;
  float lat2 = b0 * DEG2RAD;
  float dlat = (b0 - a0) * DEG2RAD;
  float dlon = (b1 - a1) * DEG2RAD;
  float s1 = __sinf(dlat * 0.5f);
  float s2 = __sinf(dlon * 0.5f);
  float a  = s1 * s1 + __cosf(lat1) * __cosf(lat2) * s2 * s2;
  float c  = 2.0f * atan2f(sqrtf(a), sqrtf(fmaxf(1.0f - a, 0.0f)));
  float d  = R_EARTH * c;
  pos[i * N + j] = (unsigned char)((d < POS_TH) && (i != j));
  neg[i * N + j] = (unsigned char)(d > NEG_TH);
}

// ---------------------------------------------------------------------------
// Kernel 3: WMMA Gram matrix -> euclidean embedding distance matrix
//   One 16x16 output tile per wave, 4 waves per block, K-loop 256/32 = 8.
//   sq[m,n] = ||e_m||^2 + ||e_n||^2 - 2*G[m,n];  dist = sqrt(max(sq,0))
// ---------------------------------------------------------------------------
__global__ void gram_dist_kernel(const _Float16* __restrict__ Eh,
                                 const float* __restrict__ norms,
                                 float* __restrict__ Dm) {
  const int wave = threadIdx.x >> 5;
  const int lane = threadIdx.x & 31;
  const int tile = blockIdx.x * 4 + wave;      // 0 .. 575
  const int ti = tile / NT, tj = tile % NT;
  const int m0 = ti * 16, n0 = tj * 16;

  // A fragment: lane row = m0 + (lane&15); two 8-half chunks at
  // K = kb + (lane>=16 ? 16:0) and K+8  (contiguous, matches B)
  const int hiA = (lane >> 4) * 16;
  // B fragment: lane col n = n0 + (lane&15); 16 contiguous halves at
  // K = kb + (lane>=16 ? 16:0).  B[k,n] = E[n,k] -> read row n of Eh.
  const int hiB = (lane >> 4) * 16;

  const _Float16* ap = Eh + (m0 + (lane & 15)) * DIM;
  const _Float16* bp = Eh + (n0 + (lane & 15)) * DIM;

  v8f c = {};
#pragma unroll
  for (int kb = 0; kb < DIM; kb += 32) {
    v8h a0 = *(const v8h*)(ap + kb + hiA);
    v8h a1 = *(const v8h*)(ap + kb + hiA + 8);
    v16h a = __builtin_shufflevector(a0, a1, 0, 1, 2, 3, 4, 5, 6, 7,
                                             8, 9, 10, 11, 12, 13, 14, 15);
    v16h b = *(const v16h*)(bp + kb + hiB);
    // (neg_a, A, neg_b, B, c_mod, C, reuse_a, reuse_b)
    c = __builtin_amdgcn_wmma_f32_16x16x32_f16(false, a, false, b,
                                               (short)0, c, false, false);
  }

  // C/D layout: VGPR v -> row v + (lane>=16 ? 8 : 0); col = lane & 15
  const int mbase = m0 + ((lane >> 4) * 8);
  const int col   = n0 + (lane & 15);
  const float nn  = norms[col];
#pragma unroll
  for (int v = 0; v < 8; ++v) {
    const int m = mbase + v;
    float sq = norms[m] + nn - 2.0f * c[v];
    Dm[m * N + col] = sqrtf(fmaxf(sq, 0.0f));
  }
}

// ---------------------------------------------------------------------------
// Kernel 4: per-anchor triplet reduction (row of D + masks live in LDS)
// ---------------------------------------------------------------------------
__global__ void triplet_kernel(const float* __restrict__ Dm,
                               const unsigned char* __restrict__ pos,
                               const unsigned char* __restrict__ neg,
                               float* __restrict__ pl,
                               unsigned int* __restrict__ pv,
                               unsigned int* __restrict__ pa) {
  const int i = blockIdx.x;
  const int t = threadIdx.x;             // 256 threads
  __shared__ float sD[N];
  __shared__ unsigned char sP[N];
  __shared__ unsigned char sN[N];
  __shared__ float rs[256];
  __shared__ unsigned int ra[256];
  __shared__ unsigned int rc[256];

  for (int idx = t; idx < N; idx += 256) {
    sD[idx] = Dm[i * N + idx];
    sP[idx] = pos[i * N + idx];
    sN[idx] = neg[i * N + idx];
  }
  __syncthreads();

  unsigned int cp = 0, cn = 0;
  for (int k = t; k < N; k += 256) { cp += sP[k]; cn += sN[k]; }

  float lsum = 0.0f;
  unsigned int lact = 0;
  for (int j = 0; j < N; ++j) {
    if (!sP[j]) continue;
    const float dap = sD[j] + MARGIN;
    for (int k = t; k < N; k += 256) {
      if (sN[k]) {
        float v = dap - sD[k];
        if (v > 0.0f) { lsum += v; lact++; }
      }
    }
  }

  rs[t] = lsum;
  ra[t] = lact;
  rc[t] = cp | (cn << 16);               // pack: both counts <= 384
  __syncthreads();
  for (int s = 128; s > 0; s >>= 1) {
    if (t < s) { rs[t] += rs[t + s]; ra[t] += ra[t + s]; rc[t] += rc[t + s]; }
    __syncthreads();
  }
  if (t == 0) {
    unsigned int np = rc[0] & 0xFFFFu;
    unsigned int nn = rc[0] >> 16;
    pl[i] = rs[0];
    pa[i] = ra[0];
    pv[i] = np * nn;
  }
}

// ---------------------------------------------------------------------------
// Kernel 5: fixed-order final reduction -> (loss, n_valid, n_active)
// ---------------------------------------------------------------------------
__global__ void finalize_kernel(const float* __restrict__ pl,
                                const unsigned int* __restrict__ pv,
                                const unsigned int* __restrict__ pa,
                                float* __restrict__ out) {
  __shared__ float fs[512];
  __shared__ unsigned int vs[512];
  __shared__ unsigned int as_[512];
  const int t = threadIdx.x;             // 512 threads
  fs[t]  = (t < N) ? pl[t] : 0.0f;
  vs[t]  = (t < N) ? pv[t] : 0u;
  as_[t] = (t < N) ? pa[t] : 0u;
  __syncthreads();
  for (int s = 256; s > 0; s >>= 1) {
    if (t < s) { fs[t] += fs[t + s]; vs[t] += vs[t + s]; as_[t] += as_[t + s]; }
    __syncthreads();
  }
  if (t == 0) {
    unsigned int nv = vs[0];
    float denom = (float)(nv > 0u ? nv : 1u);
    out[0] = fs[0] / denom;
    out[1] = (float)nv;
    out[2] = (float)as_[0];
  }
}

// ---------------------------------------------------------------------------
extern "C" void kernel_launch(void* const* d_in, const int* in_sizes, int n_in,
                              void* d_out, int out_size, void* d_ws, size_t ws_size,
                              hipStream_t stream) {
  const float* E   = (const float*)d_in[0];   // (384, 256) f32
  const float* gps = (const float*)d_in[1];   // (384, 2)   f32
  float* out = (float*)d_out;                 // 3 floats: loss, n_valid, n_active

  // workspace layout (total ~1.09 MB, L2-resident)
  char* ws = (char*)d_ws;
  _Float16*      Eh    = (_Float16*)(ws + 0);            // 196608 B
  float*         norms = (float*)(ws + 196608);          //   1536 B
  float*         Dm    = (float*)(ws + 198144);          // 589824 B
  unsigned char* pos   = (unsigned char*)(ws + 787968);  // 147456 B
  unsigned char* neg   = (unsigned char*)(ws + 935424);  // 147456 B
  float*         pl    = (float*)(ws + 1082880);         //   1536 B
  unsigned int*  pv    = (unsigned int*)(ws + 1084416);  //   1536 B
  unsigned int*  pa    = (unsigned int*)(ws + 1085952);  //   1536 B

  convert_norm_kernel<<<N, DIM, 0, stream>>>(E, Eh, norms);
  mask_kernel<<<N, N, 0, stream>>>(gps, pos, neg);
  gram_dist_kernel<<<(NT * NT) / 4, 128, 0, stream>>>(Eh, norms, Dm);
  triplet_kernel<<<N, 256, 0, stream>>>(Dm, pos, neg, pl, pv, pa);
  finalize_kernel<<<1, 512, 0, stream>>>(pl, pv, pa, out);
}
